// FineDeformableTransformerEncoderLayer_36601711296910
// MI455X (gfx1250) — compile-verified, matrix operand order checked by reading, not run on hardware
//
#include <hip/hip_runtime.h>

// ---------------- problem constants ----------------
#define D_MODEL  256
#define N_HEADS  8
#define N_LEVELS 4
#define N_POINTS 4
#define N_VT     8
#define NSEEN    64
#define D_HEAD   32
#define BATCH    2
#define L_SP     21760                  // 128^2+64^2+32^2+16^2
#define LQ       21792                  // L_SP + N_LEVELS*N_VT
#define T_VT     72                     // N_VT + NSEEN
#define ROWS     ((size_t)BATCH * LQ)   // 43584

typedef __attribute__((ext_vector_type(16))) _Float16 v16h;
typedef __attribute__((ext_vector_type(8)))  _Float16 v8h;
typedef __attribute__((ext_vector_type(8)))  float    v8f;

// =====================================================================
// Tiled WMMA GEMM:  C[M,N] = act(A[M,K] @ B[K,N] + bias[N] (+ res[M,N]))
//
// N, K are template constants -> all global-load strides fold into 24-bit
// immediate IOFFSETs (no per-load 64-bit address adds / xcnt waits).
// Block tile 64(M) x 128(N), K-step 32, 8 waves (wave32).
// Wave (wm = wid&3, wn = wid>>2) owns a 16x64 strip: 1 A fragment
// (reused) x 4 B fragments = 4 v_wmma_f32_16x16x32_f16 per K-step.
// Double-buffered LDS: next tile's global loads overlap current WMMAs,
// one workgroup barrier per K-step.
//
// LDS tiles are stored pre-swizzled into the CDNA5 wave32 fragment layout
// (CDNA5 ISA):
//   A frag (16x32 f16): lane = M%16 + 16*((K/8)&1), elem = K%8 + 8*(K/16)
//   B frag (32x16 f16): lane = N%16 + 16*((K/8)&1), elem = K%8 + 8*(K/16)
//   C/D   (16x16 f32):  lane = N%16 + 16*(M/8),     elem = M%8
// Loads are guarded by CLAMPING indices (no exec-mask divergence); the
// epilogue masks out-of-range stores. K must be a multiple of 32.
// act: 0 = none, 1 = relu, 2 = exact gelu
// =====================================================================
template<int N, int K>
__global__ __launch_bounds__(256)
void wmma_gemm_kernel(const float* __restrict__ A, const float* __restrict__ Bm,
                      const float* __restrict__ bias, const float* __restrict__ res,
                      float* __restrict__ C, int M, int act)
{
    __shared__ __align__(32) _Float16 As[2][2048];  // 4 m-groups * 32 lanes * 16 halves
    __shared__ __align__(32) _Float16 Bs[2][8192];  // 8 n-groups * 32 lanes * 16 halves

    const int tid  = threadIdx.x;
    const int lane = tid & 31;
    const int wid  = tid >> 5;
    const int wm   = wid & 3;     // m-group 0..3
    const int wn   = wid >> 2;    // 0/1 -> n-groups [4*wn, 4*wn+4)
    const int m0   = blockIdx.y << 6;
    const int n0   = blockIdx.x << 7;

    // A loader: thread -> (row r = tid>>2, cols c0..c0+7), one b128 LDS store
    const int a_r   = tid >> 2;
    const int a_c0  = (tid & 3) << 3;
    const int a_ln  = (a_r & 15) + ((tid & 1) << 4);          // (c0>>3)&1 == tid&1
    const int a_idx = ((a_r >> 4) << 9) + (a_ln << 4) + (((tid & 3) >> 1) << 3);
    int a_gr = m0 + a_r; if (a_gr >= M) a_gr = M - 1;
    const float* a_base = A + (size_t)a_gr * K + a_c0;

    // B loader: thread -> (col nn = tid&127, K-rows kk0..kk0+15), two b128 stores
    const int b_nn  = tid & 127;
    const int b_ks  = tid >> 7;                // 0/1 -> kk0 = 16*b_ks
    int b_gn = n0 + b_nn; if (b_gn >= N) b_gn = N - 1;
    const float* b_base = Bm + (size_t)(b_ks << 4) * N + b_gn;
    const int b_i0 = ((b_nn >> 4) << 9) + ((b_nn & 15) << 4) + (b_ks << 3);
    const int b_i1 = b_i0 + (16 << 4);         // half-1 lanes (+16 lanes)

    auto load_tile = [&](int k0, int p) {
        // ---- A tile 64x32 -> LDS (one b128 per thread) ----
        const float* ap = a_base + k0;
        float4 q0 = *(const float4*)ap;
        float4 q1 = *(const float4*)(ap + 4);
        v8h hv;
        hv[0] = (_Float16)q0.x; hv[1] = (_Float16)q0.y;
        hv[2] = (_Float16)q0.z; hv[3] = (_Float16)q0.w;
        hv[4] = (_Float16)q1.x; hv[5] = (_Float16)q1.y;
        hv[6] = (_Float16)q1.z; hv[7] = (_Float16)q1.w;
        *(v8h*)&As[p][a_idx] = hv;
        // ---- B tile 32x128 -> LDS (two b128 per thread) ----
        // stride N is a compile-time constant -> immediate offsets
        const float* bp = b_base + (size_t)k0 * N;
        float bv[16];
#pragma unroll
        for (int j = 0; j < 16; ++j) bv[j] = bp[j * N];
        v8h h0, h1;
#pragma unroll
        for (int j = 0; j < 8; ++j) {
            h0[j] = (_Float16)bv[j];
            h1[j] = (_Float16)bv[8 + j];
        }
        *(v8h*)&Bs[p][b_i0] = h0;
        *(v8h*)&Bs[p][b_i1] = h1;
    };

    v8f acc[4];
#pragma unroll
    for (int j = 0; j < 4; ++j) acc[j] = (v8f){0.f,0.f,0.f,0.f,0.f,0.f,0.f,0.f};

    load_tile(0, 0);
    __syncthreads();

    int p = 0;
    for (int k0 = 0; k0 < K; k0 += 32, p ^= 1) {
        if (k0 + 32 < K) load_tile(k0 + 32, p ^ 1);   // overlaps with WMMAs below
        if (k0 + 64 < K) {
            __builtin_prefetch(a_base + k0 + 64, 0, 1);
            __builtin_prefetch(b_base + (size_t)(k0 + 64) * N, 0, 1);
        }
        v16h af = *(const v16h*)&As[p][(wm << 9) + (lane << 4)];
#pragma unroll
        for (int j = 0; j < 4; ++j) {
            v16h bf = *(const v16h*)&Bs[p][(((wn << 2) + j) << 9) + (lane << 4)];
            acc[j] = __builtin_amdgcn_wmma_f32_16x16x32_f16(false, af, false, bf,
                                                            (short)0, acc[j],
                                                            false, false);
        }
        __syncthreads();
    }

    // ---- epilogue ----
    const int mb = m0 + (wm << 4) + ((lane >> 4) << 3);
#pragma unroll
    for (int j = 0; j < 4; ++j) {
        int nc = n0 + (((wn << 2) + j) << 4) + (lane & 15);
        if (nc < N) {
#pragma unroll
            for (int e = 0; e < 8; ++e) {
                int mm = mb + e;
                if (mm < M) {
                    float v = acc[j][e];
                    if (bias) v += bias[nc];
                    if (res)  v += res[(size_t)mm * N + nc];
                    if (act == 1) v = fmaxf(v, 0.f);
                    else if (act == 2)
                        v = 0.5f * v * (1.f + erff(v * 0.7071067811865476f));
                    C[(size_t)mm * N + nc] = v;
                }
            }
        }
    }
}

// =====================================================================
// q = concat(src+pos, sel_vt); vin = concat(src, sel_vt)   (B,LQ,256)
// =====================================================================
__global__ void build_q_vin_kernel(const float* __restrict__ src,
                                   const float* __restrict__ pos,
                                   const float* __restrict__ sel,
                                   float* __restrict__ q, float* __restrict__ vin)
{
    size_t i = (size_t)blockIdx.x * 256 + threadIdx.x;
    if (i >= ROWS * D_MODEL) return;
    int    col = (int)(i & 255);
    size_t row = i >> 8;                 // b*LQ + r
    size_t b   = row / LQ;
    size_t r   = row - b * LQ;
    if (r < L_SP) {
        float s = src[(b * L_SP + r) * D_MODEL + col];
        q[i]   = s + pos[(b * L_SP + r) * D_MODEL + col];
        vin[i] = s;
    } else {
        float s = sel[(b * (N_LEVELS * N_VT) + (r - L_SP)) * D_MODEL + col];
        q[i] = s; vin[i] = s;
    }
}

// =====================================================================
// softmax over last dim (48) of logits (B*LQ*H, 48), in place
// =====================================================================
__global__ void softmax48_kernel(float* __restrict__ p, size_t rows)
{
    size_t r = (size_t)blockIdx.x * 256 + threadIdx.x;
    if (r >= rows) return;
    float* v = p + r * 48;
    float mx = -1e30f;
#pragma unroll
    for (int i = 0; i < 48; ++i) mx = fmaxf(mx, v[i]);
    float s = 0.f;
#pragma unroll
    for (int i = 0; i < 48; ++i) { float e = __expf(v[i] - mx); v[i] = e; s += e; }
    float inv = 1.f / s;
#pragma unroll
    for (int i = 0; i < 48; ++i) v[i] *= inv;
}

// =====================================================================
// Deformable sampling + VT token mixing.
// One thread per (b,q,h,d). value:(B,LQ,8,32)  off:(B,LQ,8,4,4,2)
// prob:(B,LQ,8,48)  refp:(B,LQ,4,2)  ss:(4,2) int  -> out:(B,LQ,256)
// =====================================================================
__global__ void sample_kernel(const float* __restrict__ value,
                              const float* __restrict__ off,
                              const float* __restrict__ prob,
                              const float* __restrict__ refp,
                              const int*   __restrict__ ss,
                              float* __restrict__ out)
{
    size_t i = (size_t)blockIdx.x * 256 + threadIdx.x;
    if (i >= ROWS * D_MODEL) return;
    int    d  = (int)(i & 31);
    int    h  = (int)((i >> 5) & 7);
    size_t bq = i >> 8;                       // b*LQ + q
    size_t b  = bq / LQ;

    const float* ref = refp + bq * (N_LEVELS * 2);
    const float* ofs = off  + ((bq * N_HEADS + h) * N_LEVELS * N_POINTS) * 2;
    const float* pr  = prob + (bq * N_HEADS + h) * 48;

    float acc = 0.f;
    int s0 = 0;
    for (int l = 0; l < N_LEVELS; ++l) {
        int Hl = ss[2 * l], Wl = ss[2 * l + 1];
        float rx = ref[2 * l], ry = ref[2 * l + 1];
        float fW = (float)Wl, fH = (float)Hl;
        for (int p = 0; p < N_POINTS; ++p) {
            float ox = ofs[(l * N_POINTS + p) * 2];
            float oy = ofs[(l * N_POINTS + p) * 2 + 1];
            float x = (rx + ox / fW) * fW - 0.5f;
            float y = (ry + oy / fH) * fH - 0.5f;
            float x0 = floorf(x), y0 = floorf(y);
            float wpt = pr[l * 12 + p];
#pragma unroll
            for (int dy = 0; dy < 2; ++dy)
#pragma unroll
            for (int dx = 0; dx < 2; ++dx) {
                float xi = x0 + (float)dx, yi = y0 + (float)dy;
                if (xi >= 0.f && xi <= fW - 1.f && yi >= 0.f && yi <= fH - 1.f) {
                    float w = (1.f - fabsf(x - xi)) * (1.f - fabsf(y - yi)) * wpt;
                    int xc = (int)xi, yc = (int)yi;
                    size_t sp = (size_t)s0 + (size_t)yc * Wl + xc;
                    acc += w * value[((b * LQ + sp) * N_HEADS + h) * D_HEAD + d];
                }
            }
        }
#pragma unroll
        for (int v = 0; v < N_VT; ++v) {
            float w = pr[l * 12 + N_POINTS + v];
            size_t row = (size_t)L_SP + l * N_VT + v;
            acc += w * value[((b * LQ + row) * N_HEADS + h) * D_HEAD + d];
        }
        s0 += Hl * Wl;
    }
    out[i] = acc;
}

// =====================================================================
// LayerNorm over D=256, one wave per row: x = LN(x (+ add)) * g + b
// =====================================================================
__global__ __launch_bounds__(256)
void ln_kernel(float* __restrict__ x, const float* __restrict__ add,
               const float* __restrict__ g, const float* __restrict__ b, int rows)
{
    int row  = (blockIdx.x << 3) + (threadIdx.x >> 5);
    int lane = threadIdx.x & 31;
    if (row >= rows) return;
    float* xr = x + (size_t)row * D_MODEL;
    const float* ar = add ? add + (size_t)row * D_MODEL : (const float*)0;
    float v[8]; float s = 0.f;
#pragma unroll
    for (int i = 0; i < 8; ++i) {
        float t = xr[lane + (i << 5)];
        if (ar) t += ar[lane + (i << 5)];
        v[i] = t; s += t;
    }
#pragma unroll
    for (int m = 16; m >= 1; m >>= 1) s += __shfl_xor(s, m, 32);
    float mean = s * (1.f / 256.f);
    float var = 0.f;
#pragma unroll
    for (int i = 0; i < 8; ++i) { float dd = v[i] - mean; var += dd * dd; }
#pragma unroll
    for (int m = 16; m >= 1; m >>= 1) var += __shfl_xor(var, m, 32);
    float inv = rsqrtf(var * (1.f / 256.f) + 1e-5f);
#pragma unroll
    for (int i = 0; i < 8; ++i) {
        int c = lane + (i << 5);
        xr[c] = (v[i] - mean) * inv * g[c] + b[c];
    }
}

// =====================================================================
// vts (4, B, 72, 256): tokens 0..7 from x[:, L_+l*8+t], 8..71 from seen_vts
// =====================================================================
__global__ void build_vts_kernel(const float* __restrict__ x,
                                 const float* __restrict__ seen,
                                 float* __restrict__ vts)
{
    size_t i = (size_t)blockIdx.x * 256 + threadIdx.x;
    if (i >= (size_t)N_LEVELS * BATCH * T_VT * D_MODEL) return;
    int    col = (int)(i & 255);
    size_t r   = i >> 8;
    int t  = (int)(r % T_VT);
    size_t r2  = r / T_VT;
    int bb = (int)(r2 % BATCH);
    int l  = (int)(r2 / BATCH);
    float v;
    if (t < N_VT)
        v = x[((size_t)bb * LQ + L_SP + l * N_VT + t) * D_MODEL + col];
    else
        v = seen[(((size_t)bb * N_LEVELS + l) * NSEEN + (t - N_VT)) * D_MODEL + col];
    vts[i] = v;
}

// =====================================================================
// Small VT self-attention for one level: qkv (B*72, 768) -> out (B*72, 256)
// grid = B*8 blocks; thread i = query token.
// =====================================================================
__global__ __launch_bounds__(128)
void vt_attn_kernel(const float* __restrict__ qkv, float* __restrict__ out)
{
    int g = blockIdx.x;
    int h = g & 7;
    int b = g >> 3;
    const float* base = qkv + (size_t)b * T_VT * 768;
    float* ob = out + (size_t)b * T_VT * D_MODEL;
    int i = threadIdx.x;
    if (i >= T_VT) return;

    const float scale = 0.1767766952966369f;   // 32^-0.5
    float qv[D_HEAD];
#pragma unroll
    for (int d = 0; d < D_HEAD; ++d)
        qv[d] = base[(size_t)i * 768 + h * D_HEAD + d] * scale;

    float sc[T_VT]; float mx = -1e30f;
    for (int k = 0; k < T_VT; ++k) {
        const float* kr = base + (size_t)k * 768 + 256 + h * D_HEAD;
        float s = 0.f;
#pragma unroll
        for (int d = 0; d < D_HEAD; ++d) s += qv[d] * kr[d];
        sc[k] = s; mx = fmaxf(mx, s);
    }
    float ssum = 0.f;
    for (int k = 0; k < T_VT; ++k) { float e = __expf(sc[k] - mx); sc[k] = e; ssum += e; }
    float inv = 1.f / ssum;
    float o[D_HEAD];
#pragma unroll
    for (int d = 0; d < D_HEAD; ++d) o[d] = 0.f;
    for (int k = 0; k < T_VT; ++k) {
        float a = sc[k] * inv;
        const float* vr = base + (size_t)k * 768 + 512 + h * D_HEAD;
#pragma unroll
        for (int d = 0; d < D_HEAD; ++d) o[d] += a * vr[d];
    }
#pragma unroll
    for (int d = 0; d < D_HEAD; ++d)
        ob[(size_t)i * D_MODEL + h * D_HEAD + d] = o[d];
}

// =====================================================================
// Pack outputs: [x[:, :L_], seen_new (B,4,64,256), sel_new (B,32,256)]
// =====================================================================
__global__ void pack_out_kernel(const float* __restrict__ x,
                                const float* __restrict__ vts,
                                float* __restrict__ out)
{
    const size_t n1 = (size_t)BATCH * L_SP * D_MODEL;
    const size_t n2 = (size_t)BATCH * N_LEVELS * NSEEN * D_MODEL;
    const size_t n3 = (size_t)BATCH * N_LEVELS * N_VT * D_MODEL;
    size_t i = (size_t)blockIdx.x * 256 + threadIdx.x;
    if (i < n1) {
        int    col = (int)(i & 255);
        size_t row = i >> 8;
        size_t b   = row / L_SP;
        size_t r   = row - b * L_SP;
        out[i] = x[(b * LQ + r) * D_MODEL + col];
    } else if (i < n1 + n2) {
        size_t j = i - n1;
        int col = (int)(j & 255);
        size_t r = j >> 8;
        int s  = (int)(r % NSEEN); size_t r2 = r / NSEEN;
        int l  = (int)(r2 % N_LEVELS);
        int b  = (int)(r2 / N_LEVELS);
        out[i] = vts[(((size_t)l * BATCH + b) * T_VT + N_VT + s) * D_MODEL + col];
    } else if (i < n1 + n2 + n3) {
        size_t j = i - n1 - n2;
        int col = (int)(j & 255);
        size_t r = j >> 8;
        int v  = (int)(r % N_VT); size_t r2 = r / N_VT;
        int l  = (int)(r2 % N_LEVELS);
        int b  = (int)(r2 / N_LEVELS);
        out[i] = vts[(((size_t)l * BATCH + b) * T_VT + v) * D_MODEL + col];
    }
}

// =====================================================================
// host side
// =====================================================================
extern "C" void kernel_launch(void* const* d_in, const int* in_sizes, int n_in,
                              void* d_out, int out_size, void* d_ws, size_t ws_size,
                              hipStream_t stream)
{
    const float* src   = (const float*)d_in[0];
    const float* pos   = (const float*)d_in[1];
    const float* refp  = (const float*)d_in[2];
    const int*   ss    = (const int*)  d_in[3];
    // d_in[4] = level_start_index (recomputed inline)
    const float* seen  = (const float*)d_in[5];
    const float* sel   = (const float*)d_in[6];
    const float* w_off = (const float*)d_in[7];
    const float* b_off = (const float*)d_in[8];
    const float* w_attn= (const float*)d_in[9];
    const float* b_attn= (const float*)d_in[10];
    const float* w_val = (const float*)d_in[11];
    const float* b_val = (const float*)d_in[12];
    const float* w_out = (const float*)d_in[13];
    const float* b_out = (const float*)d_in[14];
    const float* g1    = (const float*)d_in[15];
    const float* be1   = (const float*)d_in[16];
    const float* w_ff1 = (const float*)d_in[17];
    const float* b_ff1 = (const float*)d_in[18];
    const float* w_ff2 = (const float*)d_in[19];
    const float* b_ff2 = (const float*)d_in[20];
    const float* g2    = (const float*)d_in[21];
    const float* be2   = (const float*)d_in[22];
    const float* vt_wqkv  = (const float*)d_in[23];
    const float* vt_bqkv  = (const float*)d_in[24];
    const float* vt_wproj = (const float*)d_in[25];
    const float* vt_bproj = (const float*)d_in[26];
    const float* vt_g3 = (const float*)d_in[27];
    const float* vt_b3 = (const float*)d_in[28];
    const float* vt_g4 = (const float*)d_in[29];
    const float* vt_b4 = (const float*)d_in[30];
    const float* vt_wfc1 = (const float*)d_in[31];
    const float* vt_bfc1 = (const float*)d_in[32];
    const float* vt_wfc2 = (const float*)d_in[33];
    const float* vt_bfc2 = (const float*)d_in[34];

    float* out = (float*)d_out;
    float* ws  = (float*)d_ws;

    // workspace layout (floats).  H reuses the q..off region after sampling.
    const size_t NE    = ROWS * D_MODEL;            // 11,157,504
    const size_t F_Q    = 0;
    const size_t F_VIN  = NE;
    const size_t F_VAL  = 2 * NE;
    const size_t F_OFFS = 3 * NE;
    const size_t F_PROB = 4 * NE;                   // ROWS*384 floats
    const size_t F_ATTN = F_PROB + ROWS * 384;
    const size_t F_X    = F_ATTN + NE;
    const size_t F_H    = 0;                        // ROWS*1024 == 4*NE, fits q..off
    const size_t F_H2   = F_ATTN;                   // attn free after out-proj
    const size_t F_VTS  = F_X + NE;
    const size_t F_QKV  = F_VTS + (size_t)N_LEVELS * BATCH * T_VT * D_MODEL;
    const size_t F_VTO  = F_QKV + (size_t)BATCH * T_VT * 768;
    const size_t F_VTH  = F_VTO + (size_t)BATCH * T_VT * D_MODEL;
    const size_t F_VTH2 = F_VTH + (size_t)BATCH * T_VT * 1024;
    const size_t F_END  = F_VTH2 + (size_t)BATCH * T_VT * D_MODEL;
    if (ws_size < F_END * sizeof(float)) return;    // insufficient scratch

    float* q    = ws + F_Q;
    float* vin  = ws + F_VIN;
    float* val  = ws + F_VAL;
    float* offs = ws + F_OFFS;
    float* prob = ws + F_PROB;
    float* attn = ws + F_ATTN;
    float* x    = ws + F_X;
    float* hbuf = ws + F_H;
    float* h2   = ws + F_H2;
    float* vts  = ws + F_VTS;
    float* qkvb = ws + F_QKV;
    float* vto  = ws + F_VTO;
    float* vth  = ws + F_VTH;
    float* vth2 = ws + F_VTH2;

    const int M  = (int)ROWS;
    // shape-dispatched WMMA GEMM (all shapes in this net are compile-time)
    auto gemm = [&](const float* A, const float* Bm, const float* bias,
                    const float* res, float* C, int m, int n, int k, int act) {
        dim3 g((n + 127) / 128, (m + 63) / 64);
        dim3 blk(256);
        if      (n ==  256 && k ==  256)
            hipLaunchKernelGGL((wmma_gemm_kernel< 256, 256>), g, blk, 0, stream,
                               A, Bm, bias, res, C, m, act);
        else if (n ==  384 && k ==  256)
            hipLaunchKernelGGL((wmma_gemm_kernel< 384, 256>), g, blk, 0, stream,
                               A, Bm, bias, res, C, m, act);
        else if (n ==  768 && k ==  256)
            hipLaunchKernelGGL((wmma_gemm_kernel< 768, 256>), g, blk, 0, stream,
                               A, Bm, bias, res, C, m, act);
        else if (n == 1024 && k ==  256)
            hipLaunchKernelGGL((wmma_gemm_kernel<1024, 256>), g, blk, 0, stream,
                               A, Bm, bias, res, C, m, act);
        else if (n ==  256 && k == 1024)
            hipLaunchKernelGGL((wmma_gemm_kernel< 256,1024>), g, blk, 0, stream,
                               A, Bm, bias, res, C, m, act);
    };

    // 1. q / vin
    {
        size_t n = ROWS * D_MODEL;
        hipLaunchKernelGGL(build_q_vin_kernel, dim3((unsigned)((n + 255) / 256)),
                           dim3(256), 0, stream, src, pos, sel, q, vin);
    }
    // 2. projections (WMMA)
    gemm(vin, w_val,  b_val,  nullptr, val,  M, 256, 256, 0);
    gemm(q,   w_off,  b_off,  nullptr, offs, M, 256, 256, 0);
    gemm(q,   w_attn, b_attn, nullptr, prob, M, 384, 256, 0);
    // 3. softmax over 48
    {
        size_t rows = ROWS * N_HEADS;
        hipLaunchKernelGGL(softmax48_kernel, dim3((unsigned)((rows + 255) / 256)),
                           dim3(256), 0, stream, prob, rows);
    }
    // 4. deformable sampling + VT mixing
    {
        size_t n = ROWS * D_MODEL;
        hipLaunchKernelGGL(sample_kernel, dim3((unsigned)((n + 255) / 256)),
                           dim3(256), 0, stream, val, offs, prob, refp, ss, attn);
    }
    // 5. out-proj + residual, LN1
    gemm(attn, w_out, b_out, vin, x, M, 256, 256, 0);
    hipLaunchKernelGGL(ln_kernel, dim3((unsigned)((ROWS + 7) / 8)), dim3(256), 0,
                       stream, x, (const float*)nullptr, g1, be1, M);
    // 6. FFN + LN2 (x = LN(x + h2))
    gemm(x, w_ff1, b_ff1, nullptr, hbuf, M, 1024, 256, 1);
    gemm(hbuf, w_ff2, b_ff2, nullptr, h2, M, 256, 1024, 0);
    hipLaunchKernelGGL(ln_kernel, dim3((unsigned)((ROWS + 7) / 8)), dim3(256), 0,
                       stream, x, h2, g2, be2, M);
    // 7. VT transformer layers (per level)
    {
        size_t n = (size_t)N_LEVELS * BATCH * T_VT * D_MODEL;
        hipLaunchKernelGGL(build_vts_kernel, dim3((unsigned)((n + 255) / 256)),
                           dim3(256), 0, stream, x, seen, vts);
    }
    const int MT = BATCH * T_VT;    // 144
    for (int l = 0; l < N_LEVELS; ++l) {
        float* vl = vts + (size_t)l * BATCH * T_VT * D_MODEL;
        gemm(vl, vt_wqkv + (size_t)l * 256 * 768, vt_bqkv + (size_t)l * 768,
             nullptr, qkvb, MT, 768, 256, 0);
        hipLaunchKernelGGL(vt_attn_kernel, dim3(BATCH * N_HEADS), dim3(128), 0,
                           stream, qkvb, vto);
        gemm(vto, vt_wproj + (size_t)l * 256 * 256, vt_bproj + (size_t)l * 256,
             vl, vl, MT, 256, 256, 0);
        hipLaunchKernelGGL(ln_kernel, dim3((MT + 7) / 8), dim3(256), 0, stream,
                           vl, (const float*)nullptr,
                           vt_g3 + (size_t)l * 256, vt_b3 + (size_t)l * 256, MT);
        gemm(vl, vt_wfc1 + (size_t)l * 256 * 1024, vt_bfc1 + (size_t)l * 1024,
             nullptr, vth, MT, 1024, 256, 2);
        gemm(vth, vt_wfc2 + (size_t)l * 1024 * 256, vt_bfc2 + (size_t)l * 256,
             nullptr, vth2, MT, 256, 1024, 0);
        hipLaunchKernelGGL(ln_kernel, dim3((MT + 7) / 8), dim3(256), 0, stream,
                           vl, vth2,
                           vt_g4 + (size_t)l * 256, vt_b4 + (size_t)l * 256, MT);
    }
    // 8. pack outputs
    {
        size_t n = (size_t)BATCH * L_SP * D_MODEL
                 + (size_t)BATCH * N_LEVELS * NSEEN * D_MODEL
                 + (size_t)BATCH * N_LEVELS * N_VT * D_MODEL;
        hipLaunchKernelGGL(pack_out_kernel, dim3((unsigned)((n + 255) / 256)),
                           dim3(256), 0, stream, x, vts, out);
    }
}
